// ChainLoss_46815143526800
// MI455X (gfx1250) — compile-verified, matrix operand order checked by reading, not run on hardware
//
#include <hip/hip_runtime.h>
#include <hip/hip_bf16.h>
#include <stdint.h>

// Problem constants (match reference)
#define Bc 32
#define Tc 250
#define Dc 3424
#define Sc 4000
#define Ec 200000
#define NT 256
#define LEAKYc 0.1f

typedef __attribute__((ext_vector_type(16))) _Float16 v16h;
typedef __attribute__((ext_vector_type(8)))  float    v8f;
typedef __attribute__((ext_vector_type(4)))  unsigned int v4u;
typedef __attribute__((ext_vector_type(8)))  int      v8i;
typedef __attribute__((ext_vector_type(4)))  int      v4i;

// ---------------------------------------------------------------------------
// Wave-0 reduction of 256 f32 partials via one V_WMMA_F32_16X16X32_F16.
// Each f32 is split losslessly into (h1,h2) f16 pair; A == ones, so
// D[m,n] = sum_k B[k,n] and sum_n D[0,n] = sum of ALL 512 B elements,
// independent of the exact lane->(k,n) mapping. Accumulation is f32.
// Call with all 32 lanes of wave 0 active (EXEC all-ones requirement).
// ---------------------------------------------------------------------------
__device__ __forceinline__ float wave0_wmma_sum256(const float* __restrict__ red) {
  const int lane = threadIdx.x & 31;
  v16h a, b;
#pragma unroll
  for (int i = 0; i < 16; ++i) a[i] = (_Float16)1.0f;
#pragma unroll
  for (int i = 0; i < 8; ++i) {
    float x = red[lane * 8 + i];
    _Float16 h1 = (_Float16)x;
    float r = x - (float)h1;
    _Float16 h2 = (_Float16)r;
    b[2 * i]     = h1;
    b[2 * i + 1] = h2;
  }
  v8f c = {};
  c = __builtin_amdgcn_wmma_f32_16x16x32_f16(false, a, false, b, (short)0, c,
                                             false, false);
  float v = c[0];
  v += __shfl_xor(v, 1);
  v += __shfl_xor(v, 2);
  v += __shfl_xor(v, 4);
  v += __shfl_xor(v, 8);
  return v;  // every lane holds the total
}

// ---------------------------------------------------------------------------
// TDM: stage one contiguous row of `nelem` f32 from global into LDS.
// 1-D tile: tensor_dim0 = tile_dim0 = nelem, tensor_dim1 = tile_dim1 = 1.
// Issue from a single wave; caller waits with s_wait_tensorcnt(0).
// 6-arg builtin form: (g0 v4u, g1 v8i, g2 v4i, g3 v4i, g4 v8i, cpol)
// ---------------------------------------------------------------------------
__device__ __forceinline__ void tdm_load_row(const float* __restrict__ gsrc,
                                             float* __restrict__ lds_dst,
                                             int nelem) {
  uint64_t ga = (uint64_t)(uintptr_t)gsrc;
  uint32_t la = (uint32_t)(uintptr_t)lds_dst;  // flat LDS addr low bits = offset
  v4u g0;
  g0[0] = 1u;                                   // count=1, user descriptor
  g0[1] = la;                                   // lds_addr
  g0[2] = (uint32_t)(ga & 0xFFFFFFFFu);         // global_addr[31:0]
  g0[3] = (uint32_t)((ga >> 32) & 0x01FFFFFFu)  // global_addr[56:32]
          | (2u << 30);                         // type = 2 ("image")
  v8i g1;
  g1[0] = (int)(2u << 16);                          // data_size = 4B
  g1[1] = (int)(((unsigned)nelem & 0xFFFFu) << 16); // tensor_dim0 lo16 @63:48
  g1[2] = (int)((((unsigned)nelem >> 16) & 0xFFFFu) // tensor_dim0 hi16
                | (1u << 16));                      // tensor_dim1 lo16 = 1
  g1[3] = (int)(((unsigned)nelem & 0xFFFFu) << 16); // tile_dim0 @127:112
  g1[4] = 1;                                        // tile_dim1 = 1, tile_dim2 = 0
  g1[5] = nelem;                                    // tensor_dim0_stride lo32
  g1[6] = 0;
  g1[7] = 0;
  v4i g2 = {0, 0, 0, 0};
  v4i g3 = {0, 0, 0, 0};
  v8i g4 = {0, 0, 0, 0, 0, 0, 0, 0};
  __builtin_amdgcn_tensor_load_to_lds(g0, g1, g2, g3, g4, 0);
}

// ---------------------------------------------------------------------------
// Prep: we[e] = exp(edge_weight[e]);  fw[s] = exp(final_logw[s])
// ---------------------------------------------------------------------------
__global__ void prep_kernel(const float* __restrict__ ew,
                            const float* __restrict__ flw,
                            float* __restrict__ we, float* __restrict__ fw) {
  int i = blockIdx.x * blockDim.x + threadIdx.x;
  if (i < Ec) we[i] = __expf(ew[i]);
  if (i < Sc) fw[i] = __expf(flw[i]);
}

// ---------------------------------------------------------------------------
// Numerator: sum over (b,t) of clip(x[b,t,target[b,t]], -30, 30).
// Single block, fixed-order tree reduction -> deterministic.
// ---------------------------------------------------------------------------
__global__ __launch_bounds__(NT) void num_kernel(const float* __restrict__ x,
                                                 const int* __restrict__ target,
                                                 float* __restrict__ num_out) {
  __shared__ float red[NT];
  const int tid = threadIdx.x;
  float acc = 0.0f;
  for (int i = tid; i < Bc * Tc; i += NT) {
    int tgt = target[i];
    float v = x[(size_t)i * Dc + tgt];
    v = fminf(fmaxf(v, -30.0f), 30.0f);
    acc += v;
  }
  red[tid] = acc;
  __syncthreads();
  for (int off = NT / 2; off > 0; off >>= 1) {
    if (tid < off) red[tid] += red[tid + off];
    __syncthreads();
  }
  if (tid == 0) num_out[0] = red[0];
}

// ---------------------------------------------------------------------------
// Denominator forward recursion, probability domain with per-frame scaling.
// One workgroup per batch element. State vectors + frame scores live in LDS.
// ---------------------------------------------------------------------------
__global__ __launch_bounds__(NT) void den_kernel(
    const float* __restrict__ x, const int* __restrict__ esrc,
    const int* __restrict__ edst, const int* __restrict__ epdf,
    const float* __restrict__ we, const float* __restrict__ fw,
    float* __restrict__ den_out) {
  __shared__ float p[Sc];      // alpha (normalized probabilities)
  __shared__ float np[Sc];     // new alpha accumulator
  __shared__ float xraw[Dc];   // raw scores for this frame
  __shared__ float xs[Dc];     // exp(x - mx)
  __shared__ float red[NT];
  __shared__ float s_tot, s_mx, s_logz;

  const int tid = threadIdx.x;
  const int b = blockIdx.x;

  for (int s = tid; s < Sc; s += NT) p[s] = (s == 0) ? 1.0f : 0.0f;
  if (tid == 0) s_logz = 0.0f;
  __syncthreads();

  const float* xb = x + (size_t)b * Tc * Dc;

  for (int t = 0; t < Tc; ++t) {
    // --- stage x[b,t,:] into LDS via TDM (wave 0 issues + waits) ---
    if (tid < 32) {
      tdm_load_row(xb + (size_t)t * Dc, xraw, Dc);
      __builtin_amdgcn_s_wait_tensorcnt(0);
    }
    __syncthreads();

    // --- mx = max_d x ---
    float m = -1e30f;
    for (int d = tid; d < Dc; d += NT) m = fmaxf(m, xraw[d]);
    red[tid] = m;
    __syncthreads();
    for (int off = NT / 2; off > 0; off >>= 1) {
      if (tid < off) red[tid] = fmaxf(red[tid], red[tid + off]);
      __syncthreads();
    }
    if (tid == 0) s_mx = red[0];
    __syncthreads();
    const float mx = s_mx;

    // --- xs = exp(x - mx); np = 0 ---
    for (int d = tid; d < Dc; d += NT) xs[d] = __expf(xraw[d] - mx);
    for (int s = tid; s < Sc; s += NT) np[s] = 0.0f;
    __syncthreads();

    // --- single-pass arc accumulation (segment-sum via ds_add_f32) ---
    for (int e = tid; e < Ec; e += NT) {
      float c = p[esrc[e]] * we[e] * xs[epdf[e]];
      atomicAdd(&np[edst[e]], c);
    }
    __syncthreads();

    // --- tot = sum_s np[s]  (partials -> one WMMA reduce on wave 0) ---
    float part = 0.0f;
    for (int s = tid; s < Sc; s += NT) part += np[s];
    red[tid] = part;
    __syncthreads();
    if (tid < 32) {
      float tw = wave0_wmma_sum256(red);
      if (tid == 0) s_tot = fmaxf(tw, 1e-30f);
    }
    __syncthreads();
    const float tot = s_tot;

    // --- leaky-HMM leak to state 0, normalize, accumulate logz ---
    const float inv = 1.0f / tot;
    for (int s = tid; s < Sc; s += NT) {
      float v = np[s];
      if (s == 0) v += LEAKYc * tot;
      p[s] = v * inv;
    }
    if (tid == 0) s_logz += logf(tot) + mx;
    __syncthreads();
  }

  // --- final = log(sum_s p[s] * exp(final_logw[s])) ---
  float part = 0.0f;
  for (int s = tid; s < Sc; s += NT) part += p[s] * fw[s];
  red[tid] = part;
  __syncthreads();
  if (tid < 32) {
    float fin = wave0_wmma_sum256(red);
    if (tid == 0) den_out[b] = s_logz + logf(fmaxf(fin, 1e-30f));
  }
}

// ---------------------------------------------------------------------------
// Finalize: deterministic serial combine, loss = -(num - den)/(B*T)
// ---------------------------------------------------------------------------
__global__ void finalize_kernel(const float* __restrict__ den_arr,
                                const float* __restrict__ num_p,
                                float* __restrict__ out) {
  if (threadIdx.x == 0 && blockIdx.x == 0) {
    float den = 0.0f;
    for (int b = 0; b < Bc; ++b) den += den_arr[b];
    out[0] = -(num_p[0] - den) / (float)(Bc * Tc);
  }
}

extern "C" void kernel_launch(void* const* d_in, const int* in_sizes, int n_in,
                              void* d_out, int out_size, void* d_ws,
                              size_t ws_size, hipStream_t stream) {
  const float* x      = (const float*)d_in[0];  // [B,T,D]
  const int*   target = (const int*)d_in[1];    // [B,T]
  const int*   esrc   = (const int*)d_in[2];    // [E]
  const int*   edst   = (const int*)d_in[3];    // [E]
  const int*   epdf   = (const int*)d_in[4];    // [E]
  const float* ew     = (const float*)d_in[5];  // [E]
  const float* flw    = (const float*)d_in[6];  // [S]

  char* ws = (char*)d_ws;
  float* we      = (float*)ws;                          // E floats
  float* fw      = (float*)(ws + (size_t)Ec * 4);       // S floats
  float* den_arr = (float*)(ws + (size_t)(Ec + Sc) * 4);// B floats
  float* num_p   = den_arr + Bc;                        // 1 float

  prep_kernel<<<(Ec + 255) / 256, 256, 0, stream>>>(ew, flw, we, fw);
  num_kernel<<<1, NT, 0, stream>>>(x, target, num_p);
  den_kernel<<<Bc, NT, 0, stream>>>(x, esrc, edst, epdf, we, fw, den_arr);
  finalize_kernel<<<1, 1, 0, stream>>>(den_arr, num_p, (float*)d_out);
}
